// RobustPrompt_I_49478023250329
// MI455X (gfx1250) — matrix operand order repaired and checked by reading, not compiled, not run on hardware
//
#include <hip/hip_runtime.h>
#include <hip/hip_bf16.h>

typedef __attribute__((ext_vector_type(16))) _Float16 v16h;
typedef __attribute__((ext_vector_type(8)))  _Float16 v8h;
typedef __attribute__((ext_vector_type(8)))  float    v8f;

#define PP     10
#define TTOK   10
#define DD     128
#define MROWS  100     // P*T
#define MPAD   112     // 7 * 16
#define MTILES 7

__device__ __forceinline__ float fast_sigmoid(float x) {
    // rcp(1 + exp(-x)); v_exp_f32 + v_rcp_f32, adequate for a 0.1 threshold
    float e = __expf(-x);
    return __builtin_amdgcn_rcpf(1.0f + e);
}

// ---------------------------------------------------------------------------
// Kernel 1: total_x = concat(tokens.reshape(P*T, D), g_x)  (pure BW copy)
// ---------------------------------------------------------------------------
__global__ __launch_bounds__(256) void concat_kernel(
    const float4* __restrict__ tokens4,   // 3200 float4
    const float4* __restrict__ gx4,       // N*32 float4
    float4* __restrict__ out4,
    long tok4, long ngx4)
{
    long i = (long)blockIdx.x * blockDim.x + threadIdx.x;
    long stride = (long)gridDim.x * blockDim.x;
    // tiny tokens region (grid always covers 3200 threads)
    if (i < tok4) out4[i] = tokens4[i];
    // bulk g_x region: clean strided b128 copy
    for (long j = i; j < ngx4; j += stride)
        out4[tok4 + j] = gx4[j];
}

// ---------------------------------------------------------------------------
// Kernel 2: inner_adj[p,t,s] = prune(sigmoid(tokens_p[t] . tokens_p[s]), 0.01)
// ---------------------------------------------------------------------------
__global__ __launch_bounds__(256) void inner_kernel(
    const float* __restrict__ tokens, float* __restrict__ out)
{
    int idx = blockIdx.x * blockDim.x + threadIdx.x;
    if (idx >= PP * TTOK * TTOK) return;
    int p = idx / (TTOK * TTOK);
    int t = (idx / TTOK) % TTOK;
    int s = idx % TTOK;
    const float* a = tokens + (size_t)(p * TTOK + t) * DD;
    const float* b = tokens + (size_t)(p * TTOK + s) * DD;
    float acc = 0.0f;
    #pragma unroll 8
    for (int d = 0; d < DD; ++d) acc += a[d] * b[d];
    float sg = fast_sigmoid(acc);
    out[idx] = (sg < 0.01f) ? 0.0f : sg;
}

// ---------------------------------------------------------------------------
// Kernel 3: cross_adj[row, n] = prune+mask(sigmoid(tokens[row] . g_x[n]))
// WMMA f32<-f16 16x16x32, K=128 in 4 chunks, M padded 100->112 (7 tiles).
// One wave per 16-column n-tile; tokens staged in LDS as f16 per block.
// ---------------------------------------------------------------------------
__global__ __launch_bounds__(256) void cross_kernel(
    const float* __restrict__ tokens,   // [100][128]
    const float* __restrict__ g_x,      // [N][128]
    const int*   __restrict__ labels,   // [N]
    float* __restrict__ cross_out,      // [100][N]
    int N)
{
    __shared__ _Float16 Alds[MPAD * DD];   // 28 KB

    // Stage tokens -> LDS as f16, zero-pad rows [100, 112)
    for (int i = threadIdx.x; i < MPAD * DD; i += blockDim.x) {
        int row = i >> 7;                       // / 128
        float v = (row < MROWS) ? tokens[i] : 0.0f;
        Alds[i] = (_Float16)v;
    }
    __syncthreads();

    const int lane    = threadIdx.x & 31;
    const int waveId  = threadIdx.x >> 5;
    const int wavesPerBlock = blockDim.x >> 5;
    const int l16     = lane & 15;
    const int halfSel = lane >> 4;            // 0: lanes 0-15, 1: lanes 16-31
    const int nTiles  = N >> 4;
    const int tileStride = gridDim.x * wavesPerBlock;

    for (int tile = blockIdx.x * wavesPerBlock + waveId; tile < nTiles;
         tile += tileStride) {
        const int n0   = tile << 4;
        const int ncol = n0 + l16;

        // Prefetch this wave's next n-tile row (speculative; dropped if OOB)
        {
            int nnext = ncol + (tileStride << 4);
            if (nnext < N)
                __builtin_prefetch(g_x + (size_t)nnext * DD, 0, 1);
        }

        // --- B fragments: 32x16 f16 per K-chunk c.
        // Lane layout (contiguous k order, same as A): half h holds K = c*32 + halfSel*16 + h,
        // column N = lane%16.  16 consecutive f32 -> 4x b128 loads.
        v16h bfrag[4];
        const float* grow = g_x + (size_t)ncol * DD;
        #pragma unroll
        for (int c = 0; c < 4; ++c) {
            const float4* src =
                reinterpret_cast<const float4*>(grow + c * 32 + halfSel * 16);
            float4 q0 = src[0], q1 = src[1], q2 = src[2], q3 = src[3];
            bfrag[c][0]  = (_Float16)q0.x;  bfrag[c][1]  = (_Float16)q0.y;
            bfrag[c][2]  = (_Float16)q0.z;  bfrag[c][3]  = (_Float16)q0.w;
            bfrag[c][4]  = (_Float16)q1.x;  bfrag[c][5]  = (_Float16)q1.y;
            bfrag[c][6]  = (_Float16)q1.z;  bfrag[c][7]  = (_Float16)q1.w;
            bfrag[c][8]  = (_Float16)q2.x;  bfrag[c][9]  = (_Float16)q2.y;
            bfrag[c][10] = (_Float16)q2.z;  bfrag[c][11] = (_Float16)q2.w;
            bfrag[c][12] = (_Float16)q3.x;  bfrag[c][13] = (_Float16)q3.y;
            bfrag[c][14] = (_Float16)q3.z;  bfrag[c][15] = (_Float16)q3.w;
        }

        const int lab10 = labels[ncol] * TTOK;   // cluster row range start

        #pragma unroll 1
        for (int m = 0; m < MTILES; ++m) {
            const int arow = m * 16 + l16;    // A: M = lane%16 in both halves

            // Preload all 4 A fragments, then 4 back-to-back WMMAs
            v16h afrag[4];
            #pragma unroll
            for (int c = 0; c < 4; ++c) {
                // A layout: halves 0-15: K = c*32 + halfSel*16 + w
                //           (contiguous, matches B)
                const v8h* ap = reinterpret_cast<const v8h*>(
                    &Alds[(size_t)arow * DD + c * 32 + halfSel * 16]);
                v8h lo = ap[0];
                v8h hi = ap[1];               // +8 halves
                afrag[c] = __builtin_shufflevector(
                    lo, hi, 0, 1, 2, 3, 4, 5, 6, 7,
                            8, 9, 10, 11, 12, 13, 14, 15);
            }
            v8f acc = {};
            #pragma unroll
            for (int c = 0; c < 4; ++c) {
                acc = __builtin_amdgcn_wmma_f32_16x16x32_f16(
                    false, afrag[c], false, bfrag[c], (short)0, acc,
                    false, false);
            }

            // Epilogue: VGPR r -> row m*16 + r + halfSel*8, col = ncol
            if (m < MTILES - 1) {
                // All 16 rows valid: unconditional coalesced stores
                #pragma unroll
                for (int r = 0; r < 8; ++r) {
                    int row = m * 16 + r + halfSel * 8;
                    float s = fast_sigmoid(acc[r]);
                    bool inClus = ((unsigned)(row - lab10) < (unsigned)TTOK);
                    float outv = (s >= 0.1f && inClus) ? s : 0.0f;
                    cross_out[(size_t)row * N + ncol] = outv;
                }
            } else if (halfSel == 0) {
                // Tail tile: only rows 96..99 exist (halfSel==0, r<4)
                #pragma unroll
                for (int r = 0; r < 4; ++r) {
                    int row = (MTILES - 1) * 16 + r;
                    float s = fast_sigmoid(acc[r]);
                    bool inClus = ((unsigned)(row - lab10) < (unsigned)TTOK);
                    float outv = (s >= 0.1f && inClus) ? s : 0.0f;
                    cross_out[(size_t)row * N + ncol] = outv;
                }
            }
        }
    }
}

// ---------------------------------------------------------------------------
extern "C" void kernel_launch(void* const* d_in, const int* in_sizes, int n_in,
                              void* d_out, int out_size, void* d_ws, size_t ws_size,
                              hipStream_t stream) {
    const float* tokens = (const float*)d_in[0];   // [10][10][128]
    const float* g_x    = (const float*)d_in[1];   // [N][128]
    const int*   labels = (const int*)d_in[2];     // [N]
    const int N = in_sizes[2];                     // 500000

    float* out = (float*)d_out;
    // Output layout (flat, return order):
    //   total_x  : (100 + N) * 128 floats
    //   inner_adj: 1000 floats
    //   cross_adj: 100 * N floats
    const long totalXElems = (long)(MROWS + N) * DD;
    float* out_total = out;
    float* out_inner = out + totalXElems;
    float* out_cross = out + totalXElems + PP * TTOK * TTOK;

    // 1) concat copy (float4 vectorized)
    {
        const long tok4 = (long)MROWS * DD / 4;       // 3200
        const long ngx4 = (long)N * DD / 4;           // 16,000,000
        concat_kernel<<<4096, 256, 0, stream>>>(
            (const float4*)tokens, (const float4*)g_x, (float4*)out_total,
            tok4, ngx4);
    }

    // 2) inner adjacency (tiny)
    {
        int n = PP * TTOK * TTOK;
        inner_kernel<<<(n + 255) / 256, 256, 0, stream>>>(tokens, out_inner);
    }

    // 3) cross adjacency via WMMA
    {
        int nTiles = N >> 4;                           // 31250
        int blocks = 1024;                             // 8192 waves
        if (blocks * 8 > nTiles) blocks = (nTiles + 7) / 8;
        if (blocks < 1) blocks = 1;
        cross_kernel<<<blocks, 256, 0, stream>>>(tokens, g_x, labels,
                                                 out_cross, N);
    }
}
